// WGRGCN_57492432224405
// MI455X (gfx1250) — matrix operand order repaired and checked by reading, not compiled, not run on hardware
//
#include <hip/hip_runtime.h>
#include <hip/hip_bf16.h>

typedef __attribute__((ext_vector_type(16))) _Float16 v16h;
typedef __attribute__((ext_vector_type(8)))  float    v8f;

#define HH   64    // hidden width == OUT
#define DIN  128   // input feature width

// ---------------- fill ----------------
__global__ void k_fill(float* p, float v, size_t n) {
    size_t i = (size_t)blockIdx.x * blockDim.x + threadIdx.x;
    size_t stride = (size_t)gridDim.x * blockDim.x;
    for (; i < n; i += stride) p[i] = v;
}

// ---------------- degree (once) ----------------
__global__ void k_deg(const int* __restrict__ dst, float* __restrict__ deg, int E) {
    int i = blockIdx.x * blockDim.x + threadIdx.x;
    int stride = gridDim.x * blockDim.x;
    for (; i < E; i += stride) atomicAdd(&deg[dst[i]], 1.0f);
}

// Build an A fragment (16x32 f16): this lane holds K = {0..7}+hi*8 and {16..23}+hi*8
// relative to `ap` which already points at row_base + k0 + hi*8.
__device__ __forceinline__ v16h load_a_frag(const float* __restrict__ ap) {
    float4 p0 = *(const float4*)(ap);
    float4 p1 = *(const float4*)(ap + 4);
    float4 p2 = *(const float4*)(ap + 16);
    float4 p3 = *(const float4*)(ap + 20);
    v16h a;
    a[0]=(_Float16)p0.x;  a[1]=(_Float16)p0.y;  a[2]=(_Float16)p0.z;  a[3]=(_Float16)p0.w;
    a[4]=(_Float16)p1.x;  a[5]=(_Float16)p1.y;  a[6]=(_Float16)p1.z;  a[7]=(_Float16)p1.w;
    a[8]=(_Float16)p2.x;  a[9]=(_Float16)p2.y;  a[10]=(_Float16)p2.z; a[11]=(_Float16)p2.w;
    a[12]=(_Float16)p3.x; a[13]=(_Float16)p3.y; a[14]=(_Float16)p3.z; a[15]=(_Float16)p3.w;
    return a;
}

// ---------------- WMMA GEMM: Y[r] = X @ W[r] (r<R) and Yroot = X @ Wroot ----
// One wave computes a 32-row x 64-col output tile (2 M-tiles x 4 N-tiles,
// 8 WMMA accumulators) for one relation. Block = 8 waves = 256 rows.
// Grid = (ceil(rows/256), R+1).
__global__ void __launch_bounds__(256)
k_gemm_wmma(const float* __restrict__ X, int ldx, int K,
            const float* __restrict__ Wrel, const float* __restrict__ Wroot,
            float* __restrict__ Yrel, float* __restrict__ Yroot,
            const float* __restrict__ bias, int R, int n) {
    // B fragments pre-swizzled: fragment (ks, c) for lane l is 16 contiguous f16
    // at ldsB[(((ks*4 + c)*32 + l) * 16)]; element e = W[(ks*32 + (l>>4)*8 + (e&7) + (e>>3)*16)*HH + c*16 + (l&15)]
    __shared__ __align__(32) _Float16 ldsB[DIN * HH];   // up to 16 KB

    int r = blockIdx.y;
    const float* Wsel = (r < R) ? (Wrel + (size_t)r * K * HH) : Wroot;
    float*       Ysel = (r < R) ? (Yrel + (size_t)r * (size_t)n * HH) : Yroot;

    for (int t = threadIdx.x; t < K * HH; t += blockDim.x) {
        int e  = t & 15;
        int l  = (t >> 4) & 31;
        int c  = (t >> 9) & 3;
        int ks = t >> 11;
        int krow = ks * 32 + (l >> 4) * 8 + (e & 7) + (e >> 3) * 16;
        int ncol = c * 16 + (l & 15);
        ldsB[t] = (_Float16)Wsel[(size_t)krow * HH + ncol];
    }
    __syncthreads();

    const int lane = threadIdx.x & 31;
    const int wave = threadIdx.x >> 5;
    const int ml   = lane & 15;
    const int hi   = lane >> 4;

    const int tiles32 = (n + 31) >> 5;
    const int tile = blockIdx.x * 8 + wave;       // wave-uniform
    if (tile >= tiles32) return;

    const int row0 = tile * 32;
    int rA0 = row0 + ml;      if (rA0 > n - 1) rA0 = n - 1;
    int rA1 = row0 + 16 + ml; if (rA1 > n - 1) rA1 = n - 1;
    const float* arow0 = X + (size_t)rA0 * ldx + hi * 8;
    const float* arow1 = X + (size_t)rA1 * ldx + hi * 8;

    const v16h* bfrag = (const v16h*)ldsB;

    v8f acc[8];
    #pragma unroll
    for (int i = 0; i < 8; ++i) acc[i] = (v8f){};

    const int nKS = K >> 5;
    for (int ks = 0; ks < nKS; ++ks) {
        v16h a0 = load_a_frag(arow0 + ks * 32);
        v16h a1 = load_a_frag(arow1 + ks * 32);
        #pragma unroll
        for (int c = 0; c < 4; ++c) {
            v16h b = bfrag[(ks * 4 + c) * 32 + lane];   // 32B contiguous -> 2x ds_load_b128
            acc[c] = __builtin_amdgcn_wmma_f32_16x16x32_f16(
                false, a0, false, b, (short)0, acc[c], false, false);
            acc[4 + c] = __builtin_amdgcn_wmma_f32_16x16x32_f16(
                false, a1, false, b, (short)0, acc[4 + c], false, false);
        }
    }

    // C/D layout: VGPR v holds (M = v + hi*8, N = c*16 + ml)
    #pragma unroll
    for (int s = 0; s < 2; ++s) {
        #pragma unroll
        for (int c = 0; c < 4; ++c) {
            float badd = bias ? bias[c * 16 + ml] : 0.0f;
            #pragma unroll
            for (int v = 0; v < 8; ++v) {
                int row = row0 + s * 16 + v + hi * 8;
                if (row < n)
                    Ysel[(size_t)row * HH + c * 16 + ml] = acc[s * 4 + c][v] + badd;
            }
        }
    }
}

// ---------------- edge gather + scatter-add (mean numerator) ----------------
__global__ void k_scatter(const float* __restrict__ hr, const int* __restrict__ src,
                          const int* __restrict__ dst, const int* __restrict__ et,
                          float* __restrict__ agg, int E, int n) {
    size_t t = (size_t)blockIdx.x * blockDim.x + threadIdx.x;
    size_t stride = (size_t)gridDim.x * blockDim.x;
    size_t total = (size_t)E * HH;
    for (; t < total; t += stride) {
        int e = (int)(t >> 6);
        int c = (int)(t & 63);
        float v = hr[((size_t)et[e] * n + src[e]) * HH + c];
        atomicAdd(&agg[(size_t)dst[e] * HH + c], v);
    }
}

// ---- out = (agg/deg if agg) + base + bias ; accumulate BN sum / sumsq ----
__global__ void __launch_bounds__(256)
k_finalize_stats(const float* __restrict__ agg, const float* __restrict__ deg,
                 const float* __restrict__ base, const float* __restrict__ bias,
                 float* __restrict__ out, float* __restrict__ stats, int n) {
    __shared__ float ls[256], ls2[256];
    size_t t = (size_t)blockIdx.x * blockDim.x + threadIdx.x;
    size_t stride = (size_t)gridDim.x * blockDim.x;   // multiple of 64
    size_t total = (size_t)n * HH;
    float s = 0.0f, s2 = 0.0f;
    for (; t < total; t += stride) {
        float v = base[t] + bias[t & 63];
        if (agg) {
            size_t row = t >> 6;
            v += agg[t] / fmaxf(deg[row], 1.0f);
        }
        out[t] = v;
        s += v; s2 += v * v;
    }
    ls[threadIdx.x] = s; ls2[threadIdx.x] = s2;
    __syncthreads();
    if (threadIdx.x < 64) {
        float a = ls[threadIdx.x] + ls[threadIdx.x + 64] + ls[threadIdx.x + 128] + ls[threadIdx.x + 192];
        float b = ls2[threadIdx.x] + ls2[threadIdx.x + 64] + ls2[threadIdx.x + 128] + ls2[threadIdx.x + 192];
        atomicAdd(&stats[threadIdx.x], a);
        atomicAdd(&stats[64 + threadIdx.x], b);
    }
}

// ---------------- batchnorm + ELU/ReLU (in place) ----------------
__global__ void k_bn_act(float* __restrict__ h, const float* __restrict__ stats,
                         const float* __restrict__ g, const float* __restrict__ b,
                         int n, int relu) {
    size_t t = (size_t)blockIdx.x * blockDim.x + threadIdx.x;
    size_t stride = (size_t)gridDim.x * blockDim.x;
    size_t total = (size_t)n * HH;
    float invn = 1.0f / (float)n;
    for (; t < total; t += stride) {
        int c = (int)(t & 63);
        float mean = stats[c] * invn;
        float var  = stats[64 + c] * invn - mean * mean;
        float y = (h[t] - mean) * rsqrtf(var + 1e-5f) * g[c] + b[c];
        h[t] = relu ? fmaxf(y, 0.0f) : (y > 0.0f ? y : expf(y) - 1.0f);
    }
}

extern "C" void kernel_launch(void* const* d_in, const int* in_sizes, int n_in,
                              void* d_out, int out_size, void* d_ws, size_t ws_size,
                              hipStream_t stream) {
    const float* x     = (const float*)d_in[0];
    const float* W0    = (const float*)d_in[1];
    const float* root0 = (const float*)d_in[2];
    const float* b0    = (const float*)d_in[3];
    const float* Wh    = (const float*)d_in[4];
    const float* rooth = (const float*)d_in[5];
    const float* bh    = (const float*)d_in[6];
    const float* gamma = (const float*)d_in[7];
    const float* beta  = (const float*)d_in[8];
    const float* m1_w  = (const float*)d_in[9];
    const float* m1_b  = (const float*)d_in[10];
    const float* mg    = (const float*)d_in[11];
    const float* mb    = (const float*)d_in[12];
    const float* m2_w  = (const float*)d_in[13];
    const float* m2_b  = (const float*)d_in[14];
    const int*   src   = (const int*)d_in[15];
    const int*   dst   = (const int*)d_in[16];
    const int*   et    = (const int*)d_in[17];

    const int N = in_sizes[0] / DIN;
    const int E = in_sizes[15];
    const int R = in_sizes[1] / (DIN * HH);

    // workspace layout (floats): hr | hroot | agg | hbuf | deg | stats
    float* ws    = (float*)d_ws;
    size_t NH    = (size_t)N * HH;
    float* hr    = ws;
    float* hroot = hr + (size_t)R * NH;
    float* agg   = hroot + NH;
    float* hbuf  = agg + NH;
    float* deg   = hbuf + NH;
    float* stats = deg + N;

    dim3 blk(256);
    int tiles32 = (N + 31) / 32;
    int gemmGX = (tiles32 + 7) / 8;

    // degree (constant across layers)
    k_fill<<<dim3(256), blk, 0, stream>>>(deg, 0.0f, (size_t)N);
    k_deg<<<dim3(1024), blk, 0, stream>>>(dst, deg, E);

    const float* hin = x;
    int ldx = DIN, K = DIN;
    for (int layer = 0; layer < 3; ++layer) {
        const float* Wr  = (layer == 0) ? W0    : Wh    + (size_t)(layer - 1) * R * HH * HH;
        const float* Wrt = (layer == 0) ? root0 : rooth + (size_t)(layer - 1) * HH * HH;
        const float* bb  = (layer == 0) ? b0    : bh    + (size_t)(layer - 1) * HH;

        k_gemm_wmma<<<dim3(gemmGX, R + 1), blk, 0, stream>>>(
            hin, ldx, K, Wr, Wrt, hr, hroot, nullptr, R, N);
        k_fill<<<dim3(2048), blk, 0, stream>>>(agg, 0.0f, NH);
        k_fill<<<dim3(1), dim3(128), 0, stream>>>(stats, 0.0f, (size_t)128);
        k_scatter<<<dim3(4096), blk, 0, stream>>>(hr, src, dst, et, agg, E, N);
        k_finalize_stats<<<dim3(2048), blk, 0, stream>>>(agg, deg, hroot, bb, hbuf, stats, N);
        k_bn_act<<<dim3(2048), blk, 0, stream>>>(hbuf, stats, gamma + layer * HH,
                                                 beta + layer * HH, N, 0);
        hin = hbuf; ldx = HH; K = HH;
    }

    // MLP head: Linear -> BN -> ReLU -> Linear(+bias) into d_out
    k_gemm_wmma<<<dim3(gemmGX, 1), blk, 0, stream>>>(
        hbuf, HH, HH, nullptr, m1_w, nullptr, hroot, nullptr, 0, N);
    k_fill<<<dim3(1), dim3(128), 0, stream>>>(stats, 0.0f, (size_t)128);
    k_finalize_stats<<<dim3(2048), blk, 0, stream>>>(nullptr, deg, hroot, m1_b, hbuf, stats, N);
    k_bn_act<<<dim3(2048), blk, 0, stream>>>(hbuf, stats, mg, mb, N, 1);
    k_gemm_wmma<<<dim3(gemmGX, 1), blk, 0, stream>>>(
        hbuf, HH, HH, nullptr, m2_w, nullptr, (float*)d_out, m2_b, 0, N);
}
